// GraphEFTModel_45079976739067
// MI455X (gfx1250) — matrix operand order repaired and checked by reading, not compiled, not run on hardware
//
#include <hip/hip_runtime.h>
#include <hip/hip_bf16.h>

// ---------------------------------------------------------------------------
// GraphEFT forward on MI455X (gfx1250): bf16 WMMA for all dense GEMMs,
// L2-resident atomic scatter for the graph-attention edge stage.
// ---------------------------------------------------------------------------

#define NN   50000
#define DD   128
#define HH   4
#define CC   32
#define LL   3
#define ETE  640000
#define EHE  320000

typedef __attribute__((ext_vector_type(16))) __bf16 v16bf;
typedef __attribute__((ext_vector_type(8)))  float  v8f;

// ---------------------------------------------------------------------------
// Weight packing: f32 [128,128] row-major (K x N) -> bf16 WMMA B-fragments.
// Fragment order: [nt(8)][kk(4)][lane(32)][elem(16)]
// ISA 16-bit B 32x16 layout: lane<16 -> col n=nt*16+lane, K=kk*32+{0..7}U{16..23}
//                            lane>=16 -> col n=nt*16+lane-16, K=kk*32+{8..15}U{24..31}
// ---------------------------------------------------------------------------
__global__ void pack_w_kernel(const float* __restrict__ W, __bf16* __restrict__ dst) {
    int t = blockIdx.x * blockDim.x + threadIdx.x;   // 0..1023
    if (t >= 1024) return;
    int lane = t & 31;
    int kk   = (t >> 5) & 3;
    int nt   = t >> 7;
    int half = lane >> 4;
    int n    = nt * 16 + (lane & 15);
    int kb   = kk * 32 + half * 8;
    __bf16* d = dst + (size_t)t * 16;
#pragma unroll
    for (int e = 0; e < 16; ++e)
        d[e] = (__bf16)W[(size_t)(kb + (e & 7) + ((e >> 3) << 4)) * DD + n];
}

// ---------------------------------------------------------------------------
// A-fragment load, ISA 16-bit A 16x32 layout:
// lane<16: row m=lane,    K = kbase + {0..7} U {16..23}
// lane>=16: row m=lane-16, K = kbase + {8..15} U {24..31}
// ---------------------------------------------------------------------------
__device__ __forceinline__ v16bf load_a_frag(const float* __restrict__ xr, int kbase, int half) {
    const float* p = xr + kbase + half * 8;
    float4 f0 = *(const float4*)(p);
    float4 f1 = *(const float4*)(p + 4);
    float4 g0 = *(const float4*)(p + 16);
    float4 g1 = *(const float4*)(p + 20);
    v16bf a;
    a[0]  = (__bf16)f0.x; a[1]  = (__bf16)f0.y; a[2]  = (__bf16)f0.z; a[3]  = (__bf16)f0.w;
    a[4]  = (__bf16)f1.x; a[5]  = (__bf16)f1.y; a[6]  = (__bf16)f1.z; a[7]  = (__bf16)f1.w;
    a[8]  = (__bf16)g0.x; a[9]  = (__bf16)g0.y; a[10] = (__bf16)g0.z; a[11] = (__bf16)g0.w;
    a[12] = (__bf16)g1.x; a[13] = (__bf16)g1.y; a[14] = (__bf16)g1.z; a[15] = (__bf16)g1.w;
    return a;
}

// One wave computes a 16(M) x [nt_begin..nt_end)*16(N) tile of X@W + bias.
__device__ __forceinline__ void wave_gemm_rowtile(
    const float* __restrict__ x_tile,      // x + mbase*128
    const __bf16* __restrict__ wmat,       // packed 128x128 (8*4*32*16 bf16)
    const float* __restrict__ bias,        // [128]
    float* __restrict__ out,               // [N,128]
    int mbase, int nt_begin, int nt_end, unsigned lane)
{
    int half = lane >> 4;
    int mrow = lane & 15;
    const float* xr = x_tile + (size_t)mrow * DD;
    v16bf a[4];
#pragma unroll
    for (int kk = 0; kk < 4; ++kk) a[kk] = load_a_frag(xr, kk * 32, half);

    for (int nt = nt_begin; nt < nt_end; ++nt) {
        v8f c = {};
#pragma unroll
        for (int kk = 0; kk < 4; ++kk) {
            v16bf b = *(const v16bf*)(wmat + ((((size_t)nt * 4 + kk) * 32 + lane) << 4));
            c = __builtin_amdgcn_wmma_f32_16x16x32_bf16(
                    false, a[kk], false, b, (short)0, c, false, false);
        }
        int n = nt * 16 + mrow;
        float bb = bias[n];
#pragma unroll
        for (int r = 0; r < 8; ++r)
            out[(size_t)(mbase + half * 8 + r) * DD + n] = c[r] + bb;
    }
}

// Fused Q/K/V/skip GEMM: 8 waves = 4 matrices x 2 column halves.
__global__ void gemm_qkvs_kernel(
    const float* __restrict__ x, const __bf16* __restrict__ wpack4,
    const float* __restrict__ bq, const float* __restrict__ bk,
    const float* __restrict__ bv, const float* __restrict__ bs,
    float* __restrict__ q, float* __restrict__ k,
    float* __restrict__ v, float* __restrict__ acc)
{
    int wave = threadIdx.x >> 5;
    unsigned lane = threadIdx.x & 31;
    int mat = wave & 3;
    int nthalf = wave >> 2;
    const __bf16* wm = wpack4 + (size_t)mat * 16384;
    const float* bias = (mat == 0) ? bq : (mat == 1) ? bk : (mat == 2) ? bv : bs;
    float* out = (mat == 0) ? q : (mat == 1) ? k : (mat == 2) ? v : acc;
    int mbase = blockIdx.x * 16;
    wave_gemm_rowtile(x + (size_t)mbase * DD, wm, bias, out, mbase,
                      nthalf * 4, nthalf * 4 + 4, lane);
}

// Single-matrix GEMM (MLP head W1): 8 waves = 8 column tiles.
__global__ void gemm_single_kernel(
    const float* __restrict__ x, const __bf16* __restrict__ wm,
    const float* __restrict__ bias, float* __restrict__ out)
{
    int wave = threadIdx.x >> 5;
    unsigned lane = threadIdx.x & 31;
    int mbase = blockIdx.x * 16;
    wave_gemm_rowtile(x + (size_t)mbase * DD, wm, bias, out, mbase, wave, wave + 1, lane);
}

// ---------------------------------------------------------------------------
// Edge-embedding encoders
// ---------------------------------------------------------------------------
__global__ void tedge_kernel(const float* __restrict__ attr, const float* __restrict__ W,
                             const float* __restrict__ b, float* __restrict__ te, int E) {
    int e = blockIdx.x * blockDim.x + threadIdx.x;
    if (e >= E) return;
    float a = attr[e];
#pragma unroll
    for (int j = 0; j < 4; ++j) te[(size_t)e * 4 + j] = a * W[j] + b[j];
}

__global__ void hedge_kernel(const float* __restrict__ attr, const float* __restrict__ W,
                             const float* __restrict__ b, float* __restrict__ he, int E) {
    int e = blockIdx.x * blockDim.x + threadIdx.x;
    if (e >= E) return;
    float a0 = attr[(size_t)e * 3 + 0], a1 = attr[(size_t)e * 3 + 1], a2 = attr[(size_t)e * 3 + 2];
#pragma unroll
    for (int j = 0; j < 4; ++j)
        he[(size_t)e * 4 + j] = a0 * W[j] + a1 * W[4 + j] + a2 * W[8 + j] + b[j];
}

// ---------------------------------------------------------------------------
// Segment-softmax attention (3 edge passes)
// ---------------------------------------------------------------------------
__global__ void init_stats_kernel(float* __restrict__ amax, float* __restrict__ den, int n) {
    int t = blockIdx.x * blockDim.x + threadIdx.x;
    if (t >= n) return;
    amax[t] = -INFINITY;
    den[t]  = 0.f;
}

__device__ __forceinline__ void atomicMaxFloat(float* addr, float val) {
    unsigned int* p = (unsigned int*)addr;
    unsigned int old = *p, assumed;
    while (__uint_as_float(old) < val) {
        assumed = old;
        old = atomicCAS(p, assumed, __float_as_uint(val));
        if (old == assumed) break;
    }
}

// Pass 1: a[e,h] = (q[dst] . (k[src]+ee)) / sqrt(C); atomic segment max.
__global__ void edge_scores_kernel(
    const int* __restrict__ src, const int* __restrict__ dst,
    const float* __restrict__ q, const float* __restrict__ k,
    const float* __restrict__ eemb, const float* __restrict__ We,
    float* __restrict__ abuf, float* __restrict__ amax, int E)
{
    int t = blockIdx.x * blockDim.x + threadIdx.x;
    if (t >= E * HH) return;
    int e = t >> 2, hh = t & 3;
    int s = src[e], d = dst[e];
    float t0 = eemb[(size_t)e * 4 + 0], t1 = eemb[(size_t)e * 4 + 1];
    float t2 = eemb[(size_t)e * 4 + 2], t3 = eemb[(size_t)e * 4 + 3];
    const float* kr = k + (size_t)s * DD + hh * CC;
    const float* qr = q + (size_t)d * DD + hh * CC;
    const float* w0 = We + hh * CC, *w1 = w0 + DD, *w2 = w1 + DD, *w3 = w2 + DD;
    float sum = 0.f;
#pragma unroll 8
    for (int c = 0; c < CC; ++c) {
        float ee = t0 * w0[c] + t1 * w1[c] + t2 * w2[c] + t3 * w3[c];
        sum += qr[c] * (kr[c] + ee);
    }
    sum *= 0.17677669529663687f;   // 1/sqrt(32)
    abuf[t] = sum;
    atomicMaxFloat(&amax[(size_t)d * HH + hh], sum);
}

// Pass 2: ex = exp(a - amax[dst]); atomic segment sum.
__global__ void edge_exp_kernel(
    const int* __restrict__ dst, float* __restrict__ abuf,
    const float* __restrict__ amax, float* __restrict__ den, int E)
{
    int t = blockIdx.x * blockDim.x + threadIdx.x;
    if (t >= E * HH) return;
    int e = t >> 2, hh = t & 3;
    int d = dst[e];
    float m = amax[(size_t)d * HH + hh];
    if (!(m > -INFINITY)) m = 0.f;     // isfinite() guard from the reference
    float ex = __expf(abuf[t] - m);
    abuf[t] = ex;
    atomicAdd(&den[(size_t)d * HH + hh], ex);
}

// Pass 3: alpha-weighted message scatter: acc[dst] += (v[src]+ee)*alpha.
__global__ void edge_msg_kernel(
    const int* __restrict__ src, const int* __restrict__ dst,
    const float* __restrict__ v, const float* __restrict__ eemb,
    const float* __restrict__ We, const float* __restrict__ abuf,
    const float* __restrict__ den, float* __restrict__ acc, int E)
{
    int t = blockIdx.x * blockDim.x + threadIdx.x;
    if (t >= E * HH) return;
    int e = t >> 2, hh = t & 3;
    int s = src[e], d = dst[e];
    float alpha = abuf[t] / (den[(size_t)d * HH + hh] + 1e-16f);
    float t0 = eemb[(size_t)e * 4 + 0], t1 = eemb[(size_t)e * 4 + 1];
    float t2 = eemb[(size_t)e * 4 + 2], t3 = eemb[(size_t)e * 4 + 3];
    const float* vr = v + (size_t)s * DD + hh * CC;
    const float* w0 = We + hh * CC, *w1 = w0 + DD, *w2 = w1 + DD, *w3 = w2 + DD;
    float* ar = acc + (size_t)d * DD + hh * CC;
#pragma unroll 8
    for (int c = 0; c < CC; ++c) {
        float ee = t0 * w0[c] + t1 * w1[c] + t2 * w2[c] + t3 * w3[c];
        atomicAdd(&ar[c], (vr[c] + ee) * alpha);
    }
}

// ---------------------------------------------------------------------------
// Per-node LayerNorm variants (block of 128 threads per node)
// ---------------------------------------------------------------------------
__device__ __forceinline__ float block_mean128(float val, float* red) {
    int f = threadIdx.x;
    red[f] = val;
    __syncthreads();
#pragma unroll
    for (int s = 64; s > 0; s >>= 1) {
        if (f < s) red[f] += red[f + s];
        __syncthreads();
    }
    float m = red[0] * (1.f / 128.f);
    __syncthreads();
    return m;
}

// combined = silu(LN(accT+accH)); h_out = h_in + combined
__global__ void combine_kernel(
    const float* __restrict__ accT, const float* __restrict__ accH,
    const float* __restrict__ hin, const float* __restrict__ g,
    const float* __restrict__ b, float* __restrict__ hout)
{
    __shared__ float red[128];
    int i = blockIdx.x, f = threadIdx.x;
    float t = accT[(size_t)i * DD + f] + accH[(size_t)i * DD + f];
    float mean = block_mean128(t, red);
    float d = t - mean;
    float var = block_mean128(d * d, red);
    float nrm = d * rsqrtf(var + 1e-5f) * g[f] + b[f];
    float sil = nrm / (1.f + __expf(-nrm));
    hout[(size_t)i * DD + f] = hin[(size_t)i * DD + f] + sil;
}

__global__ void ln_kernel(const float* __restrict__ x, const float* __restrict__ g,
                          const float* __restrict__ b, float* __restrict__ out)
{
    __shared__ float red[128];
    int i = blockIdx.x, f = threadIdx.x;
    float t = x[(size_t)i * DD + f];
    float mean = block_mean128(t, red);
    float d = t - mean;
    float var = block_mean128(d * d, red);
    out[(size_t)i * DD + f] = d * rsqrtf(var + 1e-5f) * g[f] + b[f];
}

// z = LN(silu(z1))  -- silu applied before the norm, per mlp_head
__global__ void silu_ln_kernel(const float* __restrict__ z1, const float* __restrict__ g,
                               const float* __restrict__ b, float* __restrict__ out)
{
    __shared__ float red[128];
    int i = blockIdx.x, f = threadIdx.x;
    float x = z1[(size_t)i * DD + f];
    float s = x / (1.f + __expf(-x));
    float mean = block_mean128(s, red);
    float d = s - mean;
    float var = block_mean128(d * d, red);
    out[(size_t)i * DD + f] = d * rsqrtf(var + 1e-5f) * g[f] + b[f];
}

// 128 -> 3 output projection
__global__ void proj_kernel(const float* __restrict__ z, const float* __restrict__ W2,
                            const float* __restrict__ b2, float* __restrict__ out)
{
    int i = blockIdx.x * blockDim.x + threadIdx.x;
    if (i >= NN) return;
    const float* zr = z + (size_t)i * DD;
    float s0 = b2[0], s1 = b2[1], s2 = b2[2];
#pragma unroll 8
    for (int c = 0; c < DD; ++c) {
        float zc = zr[c];
        s0 += zc * W2[c * 3 + 0];
        s1 += zc * W2[c * 3 + 1];
        s2 += zc * W2[c * 3 + 2];
    }
    out[(size_t)i * 3 + 0] = s0;
    out[(size_t)i * 3 + 1] = s1;
    out[(size_t)i * 3 + 2] = s2;
}

// ---------------------------------------------------------------------------
// Host-side orchestration
// ---------------------------------------------------------------------------
extern "C" void kernel_launch(void* const* d_in, const int* in_sizes, int n_in,
                              void* d_out, int out_size, void* d_ws, size_t ws_size,
                              hipStream_t stream) {
    (void)in_sizes; (void)n_in; (void)out_size; (void)ws_size;

    const float* x       = (const float*)d_in[0];
    const int*   eit     = (const int*)d_in[1];     // [2, ET]
    const int*   eih     = (const int*)d_in[2];     // [2, EH]
    const float* eat     = (const float*)d_in[3];
    const float* eah     = (const float*)d_in[4];
    const float* W_tedge = (const float*)d_in[5];
    const float* b_tedge = (const float*)d_in[6];
    const float* W_hedge = (const float*)d_in[7];
    const float* b_hedge = (const float*)d_in[8];
    const float* Wq_t = (const float*)d_in[9],  *bq_t = (const float*)d_in[10];
    const float* Wk_t = (const float*)d_in[11], *bk_t = (const float*)d_in[12];
    const float* Wv_t = (const float*)d_in[13], *bv_t = (const float*)d_in[14];
    const float* We_t = (const float*)d_in[15];
    const float* Ws_t = (const float*)d_in[16], *bs_t = (const float*)d_in[17];
    const float* Wq_h = (const float*)d_in[18], *bq_h = (const float*)d_in[19];
    const float* Wk_h = (const float*)d_in[20], *bk_h = (const float*)d_in[21];
    const float* Wv_h = (const float*)d_in[22], *bv_h = (const float*)d_in[23];
    const float* We_h = (const float*)d_in[24];
    const float* Ws_h = (const float*)d_in[25], *bs_h = (const float*)d_in[26];
    const float* norm_g = (const float*)d_in[27], *norm_b = (const float*)d_in[28];
    const float* fg = (const float*)d_in[29], *fb = (const float*)d_in[30];
    const float* dW1 = (const float*)d_in[31], *db1 = (const float*)d_in[32];
    const float* dg  = (const float*)d_in[33], *dbt = (const float*)d_in[34];
    const float* dW2 = (const float*)d_in[35], *db2 = (const float*)d_in[36];
    const float* sW1 = (const float*)d_in[37], *sb1 = (const float*)d_in[38];
    const float* sg  = (const float*)d_in[39], *sbt = (const float*)d_in[40];
    const float* sW2 = (const float*)d_in[41], *sb2 = (const float*)d_in[42];

    float* out = (float*)d_out;

    // ---- workspace carving -------------------------------------------------
    char* base = (char*)d_ws;
    size_t off = 0;
    auto carve = [&](size_t bytes) -> void* {
        void* p = base + off;
        off += (bytes + 255) & ~(size_t)255;
        return p;
    };
    __bf16* wpack = (__bf16*)carve((size_t)26 * 16384 * sizeof(__bf16));
    float* te   = (float*)carve((size_t)ETE * 4 * sizeof(float));
    float* he   = (float*)carve((size_t)EHE * 4 * sizeof(float));
    float* qb   = (float*)carve((size_t)NN * DD * sizeof(float));
    float* kb   = (float*)carve((size_t)NN * DD * sizeof(float));
    float* vb   = (float*)carve((size_t)NN * DD * sizeof(float));
    float* accT = (float*)carve((size_t)NN * DD * sizeof(float));
    float* accH = (float*)carve((size_t)NN * DD * sizeof(float));
    float* hA   = (float*)carve((size_t)NN * DD * sizeof(float));
    float* hB   = (float*)carve((size_t)NN * DD * sizeof(float));
    float* abuf = (float*)carve((size_t)ETE * HH * sizeof(float));
    float* amax = (float*)carve((size_t)NN * HH * sizeof(float));
    float* den  = (float*)carve((size_t)NN * HH * sizeof(float));
    // head-stage buffers reuse q/k/v (free after the conv layers)
    float* hfin = qb;
    float* z1   = kb;
    float* z2   = vb;

    // ---- one-time: pack weights to bf16 WMMA fragments ---------------------
    const float* wsrc[26];
    int m = 0;
    for (int l = 0; l < LL; ++l) {
        for (int s = 0; s < 2; ++s) {
            const float* Wq = s ? Wq_h : Wq_t;
            const float* Wk = s ? Wk_h : Wk_t;
            const float* Wv = s ? Wv_h : Wv_t;
            const float* Ws = s ? Ws_h : Ws_t;
            wsrc[m++] = Wq + (size_t)l * DD * DD;
            wsrc[m++] = Wk + (size_t)l * DD * DD;
            wsrc[m++] = Wv + (size_t)l * DD * DD;
            wsrc[m++] = Ws + (size_t)l * DD * DD;
        }
    }
    wsrc[24] = dW1;
    wsrc[25] = sW1;
    for (int i = 0; i < 26; ++i)
        pack_w_kernel<<<8, 128, 0, stream>>>(wsrc[i], wpack + (size_t)i * 16384);

    // ---- edge encoders -----------------------------------------------------
    tedge_kernel<<<(ETE + 255) / 256, 256, 0, stream>>>(eat, W_tedge, b_tedge, te, ETE);
    hedge_kernel<<<(EHE + 255) / 256, 256, 0, stream>>>(eah, W_hedge, b_hedge, he, EHE);

    // ---- message-passing layers -------------------------------------------
    const int MB = NN / 16;           // 3125 row-tiles
    const float* hin = x;
    for (int l = 0; l < LL; ++l) {
        for (int s = 0; s < 2; ++s) {
            const __bf16* wp4 = wpack + (size_t)((l * 2 + s) * 4) * 16384;
            const float* bq = (s ? bq_h : bq_t) + (size_t)l * DD;
            const float* bk = (s ? bk_h : bk_t) + (size_t)l * DD;
            const float* bv = (s ? bv_h : bv_t) + (size_t)l * DD;
            const float* bs = (s ? bs_h : bs_t) + (size_t)l * DD;
            const float* We = (s ? We_h : We_t) + (size_t)l * 4 * DD;
            float* acc = s ? accH : accT;
            const int* ei  = s ? eih : eit;
            const float* em = s ? he : te;
            int E = s ? EHE : ETE;

            // q,k,v projections + skip written straight into the accumulator
            gemm_qkvs_kernel<<<MB, 256, 0, stream>>>(hin, wp4, bq, bk, bv, bs,
                                                     qb, kb, vb, acc);
            init_stats_kernel<<<(NN * HH + 255) / 256, 256, 0, stream>>>(amax, den, NN * HH);
            int EHt = E * HH;
            edge_scores_kernel<<<(EHt + 255) / 256, 256, 0, stream>>>(
                ei, ei + E, qb, kb, em, We, abuf, amax, E);
            edge_exp_kernel<<<(EHt + 255) / 256, 256, 0, stream>>>(
                ei + E, abuf, amax, den, E);
            edge_msg_kernel<<<(EHt + 255) / 256, 256, 0, stream>>>(
                ei, ei + E, vb, em, We, abuf, den, acc, E);
        }
        float* hout = (l & 1) ? hB : hA;
        combine_kernel<<<NN, 128, 0, stream>>>(accT, accH, hin,
                                               norm_g + (size_t)l * DD,
                                               norm_b + (size_t)l * DD, hout);
        hin = hout;
    }

    // ---- final norm + MLP heads -------------------------------------------
    ln_kernel<<<NN, 128, 0, stream>>>(hin, fg, fb, hfin);

    gemm_single_kernel<<<MB, 256, 0, stream>>>(hfin, wpack + (size_t)24 * 16384, db1, z1);
    silu_ln_kernel<<<NN, 128, 0, stream>>>(z1, dg, dbt, z2);
    proj_kernel<<<(NN + 127) / 128, 128, 0, stream>>>(z2, dW2, db2, out);

    gemm_single_kernel<<<MB, 256, 0, stream>>>(hfin, wpack + (size_t)25 * 16384, sb1, z1);
    silu_ln_kernel<<<NN, 128, 0, stream>>>(z1, sg, sbt, z2);
    proj_kernel<<<(NN + 127) / 128, 128, 0, stream>>>(z2, sW2, sb2, out + (size_t)NN * 3);
}
